// DownTreeLSTM_41927470743635
// MI455X (gfx1250) — matrix operand order repaired and checked by reading, not compiled, not run on hardware
//
#include <hip/hip_runtime.h>
#include <hip/hip_bf16.h>
#include <math.h>

#define NN     8192
#define KDIM   512
#define DD     512
#define PCOLS  4096   // [iou_vis(1536) | f_vis(512) | iou_lang(1536) | f_lang(512)]
#define MAXLEV 96
#define NKT    (KDIM / 32)   // 16 k-tiles of 32
#define AS     520           // padded LDS row stride (bf16) -> bank-conflict-free b128 reads

typedef __attribute__((ext_vector_type(16))) __bf16       v16bf;
typedef __attribute__((ext_vector_type(8)))  float        v8f;
typedef __attribute__((ext_vector_type(4)))  unsigned int v4u;

union V16 { v16bf v; v4u q[2]; };

__device__ __forceinline__ float sigf(float x) { return 1.0f / (1.0f + __expf(-x)); }

// ---- pack weights straight into the WMMA B-register layout ----
// Packed flat index: ((nt*NKT + kt)*32 + lane)*16 + e
//   lane holds col n = nt*16 + (lane&15), k = kt*32 + 16*(lane>>4) + e
__global__ void pack_weights_kernel(
    const float* Wioux_v, const float* Wfx_v, const float* Wioux_l, const float* Wfx_l,
    const float* Wiouh_v, const float* Wfh_v, const float* Wiouh_l, const float* Wfh_l,
    const float* bioux_v, const float* biouh_v, const float* bfx_v, const float* bfh_v,
    const float* bioux_l, const float* biouh_l, const float* bfx_l, const float* bfh_l,
    __bf16* Wxp, __bf16* Wrp, float* bias) {
  long t = blockIdx.x * (long)blockDim.x + threadIdx.x;
  if (t >= (long)KDIM * PCOLS) return;
  int e    = (int)(t & 15);
  int lane = (int)((t >> 4) & 31);
  int kt   = (int)((t >> 9) & (NKT - 1));
  int nt   = (int)(t >> 13);
  int col  = nt * 16 + (lane & 15);
  int k    = kt * 32 + ((lane >> 4) << 4) + e;
  float wx, wr;
  if (col < 1536)      { wx = Wioux_v[k*1536 + col];      wr = Wiouh_v[k*1536 + col]; }
  else if (col < 2048) { wx = Wfx_v  [k*512 + col-1536];  wr = Wfh_v  [k*512 + col-1536]; }
  else if (col < 3584) { wx = Wioux_l[k*1536 + col-2048]; wr = Wiouh_l[k*1536 + col-2048]; }
  else                 { wx = Wfx_l  [k*512 + col-3584];  wr = Wfh_l  [k*512 + col-3584]; }
  Wxp[t] = (__bf16)wx; Wrp[t] = (__bf16)wr;
  if (kt == 0 && lane < 16 && e == 0) {   // k == 0, once per column
    float b;
    if (col < 1536)      b = bioux_v[col]      + biouh_v[col];
    else if (col < 2048) b = bfx_v[col-1536]   + bfh_v[col-1536];
    else if (col < 3584) b = bioux_l[col-2048] + biouh_l[col-2048];
    else                 b = bfx_l[col-3584]   + bfh_l[col-3584];
    bias[col] = b;
  }
}

// ---- fp32 -> bf16 inputs; zero the slot-0 h row ----
__global__ void convert_inputs_kernel(const float* X, __bf16* Xb, __bf16* hbf) {
  long t = blockIdx.x * (long)blockDim.x + threadIdx.x;
  if (t < (long)NN * KDIM) Xb[t] = (__bf16)X[t];
  if (t < DD) hbf[t] = (__bf16)0.0f;
}

// ---- K=512 inner product, 2 N-tiles per wave: A fragment reused across 2 WMMAs ----
__device__ __forceinline__ void wmma_k512_x2(const __bf16* Atile, const __bf16* Wp,
                                             int nt0, int lane, v8f& acc0, v8f& acc1) {
  const __bf16* arow = Atile + (lane & 15) * AS + ((lane >> 4) << 4);
  const __bf16* bp0  = Wp + ((long)nt0 * NKT) * 512 + lane * 16;
  const __bf16* bp1  = bp0 + (long)NKT * 512;          // adjacent N-tile's packed block
#pragma unroll 4
  for (int kt = 0; kt < NKT; ++kt) {
    V16 a, b0, b1;
    const __bf16* ap = arow + kt * 32;
    a.q[0] = *(const v4u*)(ap);          // K kb..kb+7
    a.q[1] = *(const v4u*)(ap + 8);      // K kb+8..kb+15
    const __bf16* p0 = bp0 + kt * 512;
    const __bf16* p1 = bp1 + kt * 512;
    b0.q[0] = *(const v4u*)(p0); b0.q[1] = *(const v4u*)(p0 + 8);
    b1.q[0] = *(const v4u*)(p1); b1.q[1] = *(const v4u*)(p1 + 8);
    __builtin_prefetch(p0 + 4 * 512, 0, 1);   // global_prefetch_b8, 4 k-tiles ahead
    __builtin_prefetch(p1 + 4 * 512, 0, 1);
    acc0 = __builtin_amdgcn_wmma_f32_16x16x32_bf16(false, a.v, false, b0.v,
                                                   (short)0, acc0, false, false);
    acc1 = __builtin_amdgcn_wmma_f32_16x16x32_bf16(false, a.v, false, b1.v,
                                                   (short)0, acc1, false, false);
  }
}

// ---- phase 1: Xproj = Xb @ Wx + bias  (8 waves/block, 16x256 tile per block) ----
__global__ __launch_bounds__(256) void gemm_x_kernel(const __bf16* Xb, const __bf16* Wp,
                                                     const float* bias, float* Xproj) {
  __shared__ __bf16 Atile[16 * AS];
  long m0 = (long)blockIdx.y * 16;
  {   // cooperative stage: 16 rows x 512 K (each thread copies one 64B segment)
    int row = threadIdx.x >> 4, seg = threadIdx.x & 15;
    const v4u* src = (const v4u*)(Xb + (m0 + row) * (long)KDIM + seg * 32);
    v4u* dst = (v4u*)(Atile + row * AS + seg * 32);
    dst[0] = src[0]; dst[1] = src[1]; dst[2] = src[2]; dst[3] = src[3];
  }
  __syncthreads();
  int wave = threadIdx.x >> 5, lane = threadIdx.x & 31;
  int nt0  = (blockIdx.x * 8 + wave) * 2;
  int col0 = nt0 * 16 + (lane & 15);
  v8f acc0 = {}, acc1 = {};
  wmma_k512_x2(Atile, Wp, nt0, lane, acc0, acc1);
  float bc0 = bias[col0], bc1 = bias[col0 + 16];
  int mh = (lane >> 4) << 3;
#pragma unroll
  for (int r = 0; r < 8; ++r) {
    long rowoff = (m0 + mh + r) * (long)PCOLS + col0;
    Xproj[rowoff]      = acc0[r] + bc0;
    Xproj[rowoff + 16] = acc1[r] + bc1;
  }
}

// ---- single-thread: depth per node, counting sort into level buckets ----
__global__ void build_levels_kernel(const int* parent, int* lev, int* order, int* level_off) {
  if (threadIdx.x != 0 || blockIdx.x != 0) return;
  int cnt[MAXLEV];
  for (int i = 0; i < MAXLEV; ++i) cnt[i] = 0;
  for (int i = 0; i < NN; ++i) {
    int p = parent[i];
    int lv = (p == 0) ? 0 : lev[p - 1] + 1;
    if (lv >= MAXLEV) lv = MAXLEV - 1;
    lev[i] = lv;
    cnt[lv]++;
  }
  int run = 0;
  for (int L = 0; L < MAXLEV; ++L) { level_off[L] = run; run += cnt[L]; cnt[L] = level_off[L]; }
  level_off[MAXLEV] = run;
  for (int i = 0; i < NN; ++i) order[cnt[lev[i]]++] = i;
}

// ---- per level: Xproj[rows of level] += h_bf16[parent] @ Wr ----
__global__ __launch_bounds__(256) void gemm_level_kernel(int L, const int* parent, const int* order,
                                                         const int* level_off, const __bf16* hbf,
                                                         const __bf16* Wp, float* Xproj) {
  int off = level_off[L], end = level_off[L + 1];
  int ty = blockIdx.y;
  if (ty * 16 >= end - off) return;          // block-uniform exit (EXEC all-1 at WMMA)
  __shared__ __bf16 Atile[16 * AS];
  {   // cooperative gather of 16 parent-h rows into LDS
    int row = threadIdx.x >> 4, seg = threadIdx.x & 15;
    int r = off + ty * 16 + row;
    long arow = (r < end) ? (long)parent[order[r]] : 0;   // slot 0 is the zero row
    const v4u* src = (const v4u*)(hbf + arow * (long)KDIM + seg * 32);
    v4u* dst = (v4u*)(Atile + row * AS + seg * 32);
    dst[0] = src[0]; dst[1] = src[1]; dst[2] = src[2]; dst[3] = src[3];
  }
  __syncthreads();
  int wave = threadIdx.x >> 5, lane = threadIdx.x & 31;
  int nt0  = (blockIdx.x * 8 + wave) * 2;
  int col0 = nt0 * 16 + (lane & 15);
  int mh = (lane >> 4) << 3;
  long crow[8];
  v8f acc0, acc1;
#pragma unroll
  for (int r = 0; r < 8; ++r) {   // C init from stored pre-activations
    int rr = off + ty * 16 + mh + r;
    if (rr < end) {
      long n = order[rr]; crow[r] = n;
      long rowoff = n * (long)PCOLS + col0;
      acc0[r] = Xproj[rowoff]; acc1[r] = Xproj[rowoff + 16];
    } else { crow[r] = -1; acc0[r] = 0.0f; acc1[r] = 0.0f; }
  }
  wmma_k512_x2(Atile, Wp, nt0, lane, acc0, acc1);
#pragma unroll
  for (int r = 0; r < 8; ++r)
    if (crow[r] >= 0) {
      long rowoff = crow[r] * (long)PCOLS + col0;
      Xproj[rowoff]      = acc0[r];
      Xproj[rowoff + 16] = acc1[r];
    }
}

// ---- per level: LSTM cell + type-mask mix; write c/h (fp32 out) and bf16 h for next level ----
__global__ void cell_level_kernel(int L, const int* parent, const int* order, const int* level_off,
                                  const float* Xproj, const float* tmask,
                                  float* c_out, float* h_out, __bf16* hbf) {
  int off = level_off[L], end = level_off[L + 1];
  long total = (long)(end - off) * DD;
  for (long t = blockIdx.x * (long)blockDim.x + threadIdx.x; t < total;
       t += (long)gridDim.x * blockDim.x) {
    int r = off + (int)(t / DD);
    int d = (int)(t % DD);
    int n = order[r];
    int p = parent[n];
    const float* row = Xproj + (long)n * PCOLS;
    float c_p = (p == 0) ? 0.0f : c_out[(long)(p - 1) * DD + d];
    float cv = sigf(row[d]) * tanhf(row[2 * DD + d]) + sigf(row[3 * DD + d]) * c_p;
    float hv = sigf(row[DD + d]) * tanhf(cv);
    float cl = sigf(row[4 * DD + d]) * tanhf(row[6 * DD + d]) + sigf(row[7 * DD + d]) * c_p;
    float hl = sigf(row[5 * DD + d]) * tanhf(cl);
    float tm0 = tmask[n * 2], tm1 = tmask[n * 2 + 1];
    float cn = tm0 * cl + tm1 * cv;
    float hn = tm0 * hl + tm1 * hv;
    c_out[(long)n * DD + d] = cn;
    h_out[(long)n * DD + d] = hn;
    hbf[(long)(n + 1) * DD + d] = (__bf16)hn;
  }
}

extern "C" void kernel_launch(void* const* d_in, const int* in_sizes, int n_in,
                              void* d_out, int out_size, void* d_ws, size_t ws_size,
                              hipStream_t stream) {
  (void)in_sizes; (void)n_in; (void)out_size; (void)ws_size;
  const float* inputs   = (const float*)d_in[0];
  const float* tmask    = (const float*)d_in[1];
  const int*   parent   = (const int*)  d_in[2];
  const float* W_ioux_v = (const float*)d_in[3];  const float* b_ioux_v = (const float*)d_in[4];
  const float* W_iouh_v = (const float*)d_in[5];  const float* b_iouh_v = (const float*)d_in[6];
  const float* W_fx_v   = (const float*)d_in[7];  const float* b_fx_v   = (const float*)d_in[8];
  const float* W_fh_v   = (const float*)d_in[9];  const float* b_fh_v   = (const float*)d_in[10];
  const float* W_ioux_l = (const float*)d_in[11]; const float* b_ioux_l = (const float*)d_in[12];
  const float* W_iouh_l = (const float*)d_in[13]; const float* b_iouh_l = (const float*)d_in[14];
  const float* W_fx_l   = (const float*)d_in[15]; const float* b_fx_l   = (const float*)d_in[16];
  const float* W_fh_l   = (const float*)d_in[17]; const float* b_fh_l   = (const float*)d_in[18];

  char* base = (char*)d_ws;
  size_t off = 0;
  auto take = [&](size_t bytes) { char* p = base + off; off = (off + bytes + 255) & ~(size_t)255; return p; };
  __bf16* Wxp    = (__bf16*)take((size_t)KDIM * PCOLS * 2);
  __bf16* Wrp    = (__bf16*)take((size_t)KDIM * PCOLS * 2);
  float*  bias   = (float*) take((size_t)PCOLS * 4);
  __bf16* Xb     = (__bf16*)take((size_t)NN * KDIM * 2);
  __bf16* hbf    = (__bf16*)take((size_t)(NN + 1) * DD * 2);
  int*    lev    = (int*)   take((size_t)NN * 4);
  int*    order  = (int*)   take((size_t)NN * 4);
  int*    lvoff  = (int*)   take((size_t)(MAXLEV + 1) * 4);
  float*  Xproj  = (float*) take((size_t)NN * PCOLS * 4);

  float* c_out = (float*)d_out;
  float* h_out = c_out + (long)NN * DD;

  pack_weights_kernel<<<(KDIM * PCOLS + 255) / 256, 256, 0, stream>>>(
      W_ioux_v, W_fx_v, W_ioux_l, W_fx_l, W_iouh_v, W_fh_v, W_iouh_l, W_fh_l,
      b_ioux_v, b_iouh_v, b_fx_v, b_fh_v, b_ioux_l, b_iouh_l, b_fx_l, b_fh_l,
      Wxp, Wrp, bias);
  convert_inputs_kernel<<<(NN * KDIM + 255) / 256, 256, 0, stream>>>(inputs, Xb, hbf);
  gemm_x_kernel<<<dim3(PCOLS / 256, NN / 16), 256, 0, stream>>>(Xb, Wxp, bias, Xproj);
  build_levels_kernel<<<1, 32, 0, stream>>>(parent, lev, order, lvoff);
  for (int L = 0; L < MAXLEV; ++L) {
    gemm_level_kernel<<<dim3(PCOLS / 256, NN / 16), 256, 0, stream>>>(
        L, parent, order, lvoff, hbf, Wrp, Xproj);
    cell_level_kernel<<<2048, 256, 0, stream>>>(
        L, parent, order, lvoff, Xproj, tmask, c_out, h_out, hbf);
  }
}
